// SupportMemoryConditionedVelocityFieldV2_15934328668506
// MI455X (gfx1250) — compile-verified, matrix operand order checked by reading, not run on hardware
//
#include <hip/hip_runtime.h>
#include <math.h>

// ---------------- types for CDNA5 WMMA ----------------
typedef __attribute__((ext_vector_type(16))) __bf16 v16bf;
typedef __attribute__((ext_vector_type(8)))  __bf16 v8bf;
typedef __attribute__((ext_vector_type(8)))  float  v8f;
typedef __attribute__((ext_vector_type(4)))  int    v4i;

#define B_ROWS   4096
#define M_MEM    64
#define LATENT   512
#define CTX      512
#define HID      1024
#define TDIM     128
#define HEADS    8
#define HEAD_DIM 128
#define ATT_SCALE 0.08838834764831845f   // 128^-0.5

// ---- CDNA5 async global->LDS (gfx1250). Guarded; falls back to sync copy. ----
// Probe round 2: param 0 is 'v4i __device__*' (global AS, non-const). Assume
// param 1 is the LDS-side v4i pointer (shared AS), then imm offset, imm cpol.
#if __has_builtin(__builtin_amdgcn_global_load_async_to_lds_b128)
#define HAVE_ASYNC_LDS 1
typedef __attribute__((address_space(1))) v4i as1_v4i;
typedef __attribute__((address_space(3))) v4i as3_v4i;
__device__ inline void async_copy16(const __bf16* g, __bf16* l) {
  __builtin_amdgcn_global_load_async_to_lds_b128(
      (as1_v4i*)(__bf16*)g, (as3_v4i*)l, 0, 0);
}
__device__ inline void wait_async() {
#if __has_builtin(__builtin_amdgcn_s_wait_asynccnt)
  __builtin_amdgcn_s_wait_asynccnt(0);
#else
  asm volatile("s_wait_asynccnt 0" ::: "memory");
#endif
}
#else
#define HAVE_ASYNC_LDS 0
#endif

// Build a 16-element bf16 WMMA fragment from two contiguous 8-element chunks.
// Per ISA 7.12.2 (16-bit A 16x32): lane<16 holds K=kb..kb+7 (v[0..7]) and
// K=16+kb..16+kb+7 (v[8..15]) with kb = 8*(lane>=16). B assumed symmetric
// (lane&15 = column, same K split).
__device__ inline v16bf frag_load(const __bf16* lo, const __bf16* hi) {
  v8bf a = *(const v8bf*)lo;
  v8bf b = *(const v8bf*)hi;
  v16bf r;
#pragma unroll
  for (int i = 0; i < 8; ++i) { r[i] = a[i]; r[i + 8] = b[i]; }
  return r;
}

__device__ inline float gelu_exact(float x) {
  return 0.5f * x * (1.0f + erff(x * 0.70710678118654752f));
}

__device__ inline v8f wmma_bf16(v16bf a, v16bf b, v8f c) {
  return __builtin_amdgcn_wmma_f32_16x16x32_bf16(false, a, false, b, (short)0, c, false, false);
}

// ---------------- generic bf16 WMMA GEMM ----------------
// C[M,N] = epilogue(A[M,K]_bf16 @ B[K,N]_bf16 + bias) (+res), write fp32/bf16.
// Block tile 128x64, BK=32, 256 threads = 8 wave32 waves in a 4x2 grid;
// each wave owns a 32x32 region: 2 A-frags x 2 B-frags -> 4 WMMAs / K-step.
// M%128==0, N%64==0, K%32==0 guaranteed by the problem sizes.
__global__ __launch_bounds__(256) void gemm_bf16_kernel(
    const __bf16* __restrict__ A, const __bf16* __restrict__ B,
    const float* __restrict__ bias, const float* __restrict__ res,
    float* __restrict__ Cf, __bf16* __restrict__ Cb,
    int M, int N, int K, int gelu_flag)
{
  __shared__ __bf16 As[128][32];  // A tile, row-major [m][k]  (8 KB)
  __shared__ __bf16 Bs[64][32];   // B tile transposed [n][k]  (4 KB)

  const int m0 = blockIdx.y * 128;
  const int n0 = blockIdx.x * 64;   // N fastest -> consecutive blocks share A tile in L2
  const int t    = threadIdx.x;
  const int lane = t & 31;
  const int wave = t >> 5;
  const int wr = (wave & 3) * 32;   // wave row offset in tile
  const int wc = (wave >> 2) * 32;  // wave col offset in tile
  const int lrow = lane & 15;
  const int kb   = (lane >> 4) * 8;

  v8f acc00 = {0.f,0.f,0.f,0.f,0.f,0.f,0.f,0.f};
  v8f acc01 = acc00, acc10 = acc00, acc11 = acc00;

  const int ar  = t >> 2;          // 0..63 : A tile row (this thread also does ar+64)
  const int akc = (t & 3) * 8;     // A tile k-chunk (16B)
  const int bn  = t & 63;          // B tile column (coalesced in n)
  const int bkk = (t >> 6) * 8;    // B tile k-chunk

  for (int k0 = 0; k0 < K; k0 += 32) {
    // --- A tile: 16B per lane, two rows per thread, via async DMA to LDS ---
    const __bf16* ga0 = &A[(size_t)(m0 + ar) * K + (k0 + akc)];
    const __bf16* ga1 = &A[(size_t)(m0 + ar + 64) * K + (k0 + akc)];
#if HAVE_ASYNC_LDS
    async_copy16(ga0, &As[ar][akc]);
    async_copy16(ga1, &As[ar + 64][akc]);
#else
    *(v8bf*)&As[ar][akc]      = *(const v8bf*)ga0;
    *(v8bf*)&As[ar + 64][akc] = *(const v8bf*)ga1;
#endif
    // --- prefetch next B tile region while this tile is gathered/consumed ---
    if (k0 + 32 < K)
      __builtin_prefetch(&B[(size_t)(k0 + 32 + bkk) * N + (n0 + bn)], 0, 0);
    // --- B tile: transposing gather (coalesced in n across lanes) ---
#pragma unroll
    for (int j = 0; j < 8; ++j)
      Bs[bn][bkk + j] = B[(size_t)(k0 + bkk + j) * N + (n0 + bn)];
#if HAVE_ASYNC_LDS
    wait_async();
#endif
    __syncthreads();

    v16bf a0 = frag_load(&As[wr + lrow][kb],      &As[wr + lrow][16 + kb]);
    v16bf a1 = frag_load(&As[wr + 16 + lrow][kb], &As[wr + 16 + lrow][16 + kb]);
    v16bf b0 = frag_load(&Bs[wc + lrow][kb],      &Bs[wc + lrow][16 + kb]);
    v16bf b1 = frag_load(&Bs[wc + 16 + lrow][kb], &Bs[wc + 16 + lrow][16 + kb]);
    acc00 = wmma_bf16(a0, b0, acc00);
    acc01 = wmma_bf16(a0, b1, acc01);
    acc10 = wmma_bf16(a1, b0, acc10);
    acc11 = wmma_bf16(a1, b1, acc11);
    __syncthreads();
  }

  // C/D layout (ISA 7.12.2): element r -> row = r + 8*(lane>=16), col = lane&15
  const int rb0 = m0 + wr + (lane >> 4) * 8;       // A-frag 0 rows
  const int rb1 = rb0 + 16;                        // A-frag 1 rows
  const int c0 = n0 + wc + (lane & 15);
  const int c1 = c0 + 16;
  const v8f* accs[2][2] = { {&acc00, &acc01}, {&acc10, &acc11} };
  const int rbs[2] = { rb0, rb1 };
#pragma unroll
  for (int i = 0; i < 2; ++i) {
#pragma unroll
    for (int r = 0; r < 8; ++r) {
      const int rg = rbs[i] + r;
      float x0 = (*accs[i][0])[r], x1 = (*accs[i][1])[r];
      if (bias) { x0 += bias[c0]; x1 += bias[c1]; }
      if (gelu_flag) { x0 = gelu_exact(x0); x1 = gelu_exact(x1); }
      if (res) { x0 += res[(size_t)rg * N + c0]; x1 += res[(size_t)rg * N + c1]; }
      if (Cf) { Cf[(size_t)rg * N + c0] = x0; Cf[(size_t)rg * N + c1] = x1; }
      if (Cb) { Cb[(size_t)rg * N + c0] = (__bf16)x0; Cb[(size_t)rg * N + c1] = (__bf16)x1; }
    }
  }
}

// ---------------- elementwise cast fp32 -> bf16 ----------------
__global__ void cast_f32_to_bf16(const float* __restrict__ in, __bf16* __restrict__ out, size_t n) {
  size_t i = (size_t)blockIdx.x * blockDim.x + threadIdx.x;
  size_t stride = (size_t)gridDim.x * blockDim.x;
  for (; i < n; i += stride) out[i] = (__bf16)in[i];
}

// ---------------- folded bias: bk' = mem_b @ wk + bk ; bv' analogous ----------------
__global__ void fold_bias_kernel(const float* __restrict__ mem_b,
                                 const float* __restrict__ wk, const float* __restrict__ bk,
                                 const float* __restrict__ wv, const float* __restrict__ bv,
                                 float* __restrict__ bkp, float* __restrict__ bvp) {
  int n = blockIdx.x * blockDim.x + threadIdx.x;
  if (n >= HID) return;
  float sk = bk[n], sv = bv[n];
  for (int j = 0; j < HID; ++j) {
    float m = mem_b[j];
    sk += m * wk[(size_t)j * HID + n];
    sv += m * wv[(size_t)j * HID + n];
  }
  bkp[n] = sk; bvp[n] = sv;
}

// ---------------- block reduce helper (256 threads) ----------------
__device__ inline void block_reduce2(float& s, float& ss, float* rA, float* rB) {
  int t = threadIdx.x;
  rA[t] = s; rB[t] = ss; __syncthreads();
  for (int off = 128; off > 0; off >>= 1) {
    if (t < off) { rA[t] += rA[t + off]; rB[t] += rB[t + off]; }
    __syncthreads();
  }
  s = rA[0]; ss = rB[0]; __syncthreads();
}

// ---------------- trunk input: concat(states, time_embed) -> LN -> bf16 ----------------
__global__ __launch_bounds__(256) void trunk_input_kernel(
    const float* __restrict__ states, const float* __restrict__ tvals,
    const float* __restrict__ g, const float* __restrict__ b,
    __bf16* __restrict__ out)   // [B, 640]
{
  const int C = LATENT + TDIM;  // 640
  __shared__ float x[LATENT + TDIM];
  __shared__ float rA[256], rB[256];
  const int row = blockIdx.x;
  const int t = threadIdx.x;
  x[t]       = states[(size_t)row * LATENT + t];
  x[t + 256] = states[(size_t)row * LATENT + t + 256];
  if (t < TDIM / 2) {  // 64
    float e = (float)t * (1.0f / 63.0f);                    // linspace(0,1,64)
    float freq = __expf(6.907755278982137f * e);            // exp(ln(1000)*e)
    float ang = 2.0f * 3.14159265358979323846f * tvals[row] * freq;
    x[LATENT + t]            = sinf(ang);
    x[LATENT + TDIM/2 + t]   = cosf(ang);
  }
  __syncthreads();
  float s = 0.f, ss = 0.f;
  for (int c = t; c < C; c += 256) { float v = x[c]; s += v; ss += v * v; }
  block_reduce2(s, ss, rA, rB);
  float mean = s / C;
  float rs = rsqrtf(ss / C - mean * mean + 1e-5f);
  for (int c = t; c < C; c += 256)
    out[(size_t)row * C + c] = (__bf16)((x[c] - mean) * rs * g[c] + b[c]);
}

// ---------------- wave-per-row LN (fp32 in, bf16 out) ----------------
__global__ __launch_bounds__(256) void ln_rows_wave_kernel(
    const float* __restrict__ in, const float* __restrict__ g, const float* __restrict__ b,
    __bf16* __restrict__ out, int C, long nrows)
{
  long row = (long)blockIdx.x * 8 + (threadIdx.x >> 5);
  if (row >= nrows) return;
  int lane = threadIdx.x & 31;
  const float* p = in + (size_t)row * C;
  float s = 0.f, ss = 0.f;
  for (int c = lane; c < C; c += 32) { float v = p[c]; s += v; ss += v * v; }
  for (int off = 16; off > 0; off >>= 1) {
    s  += __shfl_xor(s, off, 32);
    ss += __shfl_xor(ss, off, 32);
  }
  float mean = s / C;
  float rs = rsqrtf(ss / C - mean * mean + 1e-5f);
  __bf16* q = out + (size_t)row * C;
  for (int c = lane; c < C; c += 32)
    q[c] = (__bf16)((p[c] - mean) * rs * g[c] + b[c]);
}

// ---------------- attention: per (b,head) softmax over M=64 keys ----------------
__global__ __launch_bounds__(128) void attention_kernel(
    const float* __restrict__ Q,        // [B, HID]
    const __bf16* __restrict__ Kb,      // [B*M, HID]
    const __bf16* __restrict__ Vb,      // [B*M, HID]
    __bf16* __restrict__ Sb)            // [B, HID] bf16 summary
{
  const int bh = blockIdx.x;
  const int b = bh >> 3, h = bh & 7;
  __shared__ float q[HEAD_DIM];
  __shared__ float att[M_MEM];
  __shared__ float red[M_MEM];
  const int t = threadIdx.x;

  q[t] = Q[(size_t)b * HID + h * HEAD_DIM + t];
  __syncthreads();

  if (t < M_MEM) {
    const __bf16* kp = Kb + ((size_t)(b * M_MEM + t)) * HID + h * HEAD_DIM;
    float s = 0.f;
#pragma unroll 4
    for (int d = 0; d < HEAD_DIM; ++d) s += q[d] * (float)kp[d];
    att[t] = s * ATT_SCALE;
    red[t] = att[t];
  }
  __syncthreads();
  for (int off = 32; off > 0; off >>= 1) {
    if (t < off) red[t] = fmaxf(red[t], red[t + off]);
    __syncthreads();
  }
  float mx = red[0];
  __syncthreads();
  if (t < M_MEM) { float e = __expf(att[t] - mx); att[t] = e; red[t] = e; }
  __syncthreads();
  for (int off = 32; off > 0; off >>= 1) {
    if (t < off) red[t] += red[t + off];
    __syncthreads();
  }
  float inv = 1.0f / red[0];
  __syncthreads();
  if (t < M_MEM) att[t] *= inv;
  __syncthreads();

  float acc = 0.f;
  const __bf16* vp = Vb + ((size_t)b * M_MEM) * HID + h * HEAD_DIM + t;
#pragma unroll 4
  for (int m = 0; m < M_MEM; ++m) acc += att[m] * (float)vp[(size_t)m * HID];
  Sb[(size_t)b * HID + h * HEAD_DIM + t] = (__bf16)acc;
}

// ---------------- residual + preLN + FiLM + condLN fused ----------------
__global__ __launch_bounds__(256) void film_fuse_kernel(
    const float* __restrict__ G, const float* __restrict__ ATT,
    const float* __restrict__ FILM,                      // [B, 2*HID]
    const float* __restrict__ plg, const float* __restrict__ plb,
    const float* __restrict__ clg, const float* __restrict__ clb,
    float* __restrict__ COMB, __bf16* __restrict__ CX)   // [B, HID]
{
  __shared__ float y[HID];
  __shared__ float rA[256], rB[256];
  const int row = blockIdx.x;
  const int t = threadIdx.x;
  float s = 0.f, ss = 0.f;
  for (int c = t; c < HID; c += 256) {
    float v = G[(size_t)row * HID + c] + ATT[(size_t)row * HID + c];
    y[c] = v; s += v; ss += v * v;
  }
  block_reduce2(s, ss, rA, rB);
  float mean = s / HID;
  float rs = rsqrtf(ss / HID - mean * mean + 1e-5f);
  s = 0.f; ss = 0.f;
  for (int c = t; c < HID; c += 256) {
    float v = (y[c] - mean) * rs * plg[c] + plb[c];
    float sc = FILM[(size_t)row * (2 * HID) + c];
    float sh = FILM[(size_t)row * (2 * HID) + HID + c];
    v = (1.0f + sc) * v + sh;
    COMB[(size_t)row * HID + c] = v;
    y[c] = v; s += v; ss += v * v;
  }
  block_reduce2(s, ss, rA, rB);
  float mean2 = s / HID;
  float rs2 = rsqrtf(ss / HID - mean2 * mean2 + 1e-5f);
  for (int c = t; c < HID; c += 256)
    CX[(size_t)row * HID + c] = (__bf16)((y[c] - mean2) * rs2 * clg[c] + clb[c]);
}

// ---------------- host side ----------------
static inline void* bump(char*& p, size_t bytes) {
  void* r = p;
  p += (bytes + 255) & ~(size_t)255;
  return r;
}

static inline void gemm(hipStream_t s, const __bf16* A, const __bf16* B,
                        const float* bias, const float* res,
                        float* Cf, __bf16* Cb, int M, int N, int K, int gelu) {
  dim3 grid(N / 64, M / 128);
  gemm_bf16_kernel<<<grid, 256, 0, s>>>(A, B, bias, res, Cf, Cb, M, N, K, gelu);
}

static inline void cast_bf16(hipStream_t s, const float* in, __bf16* out, size_t n) {
  int blocks = (int)((n + 1023) / 1024);
  if (blocks > 2048) blocks = 2048;
  cast_f32_to_bf16<<<blocks, 256, 0, s>>>(in, out, n);
}

extern "C" void kernel_launch(void* const* d_in, const int* in_sizes, int n_in,
                              void* d_out, int out_size, void* d_ws, size_t ws_size,
                              hipStream_t stream) {
  (void)in_sizes; (void)n_in; (void)out_size; (void)ws_size;

  const float* states   = (const float*)d_in[0];
  const float* tvals    = (const float*)d_in[1];
  const float* csum     = (const float*)d_in[2];
  const float* smem     = (const float*)d_in[3];
  const float* trunk_ln_g = (const float*)d_in[4];
  const float* trunk_ln_b = (const float*)d_in[5];
  const float* trunk_w1 = (const float*)d_in[6];
  const float* trunk_b1 = (const float*)d_in[7];
  const float* trunk_w2 = (const float*)d_in[8];
  const float* trunk_b2 = (const float*)d_in[9];
  const float* mem_ln_g = (const float*)d_in[10];
  const float* mem_ln_b = (const float*)d_in[11];
  const float* mem_w    = (const float*)d_in[12];
  const float* mem_b    = (const float*)d_in[13];
  const float* wq = (const float*)d_in[14];
  const float* bq = (const float*)d_in[15];
  const float* wk = (const float*)d_in[16];
  const float* bk = (const float*)d_in[17];
  const float* wv = (const float*)d_in[18];
  const float* bv = (const float*)d_in[19];
  const float* wo = (const float*)d_in[20];
  const float* bo = (const float*)d_in[21];
  const float* pre_ln_g = (const float*)d_in[22];
  const float* pre_ln_b = (const float*)d_in[23];
  const float* film_w = (const float*)d_in[24];
  const float* film_b = (const float*)d_in[25];
  const float* cond_ln_g = (const float*)d_in[26];
  const float* cond_ln_b = (const float*)d_in[27];
  const float* cond_w1 = (const float*)d_in[28];
  const float* cond_b1 = (const float*)d_in[29];
  const float* cond_w2 = (const float*)d_in[30];
  const float* cond_b2 = (const float*)d_in[31];
  const float* head_ln_g = (const float*)d_in[32];
  const float* head_ln_b = (const float*)d_in[33];
  const float* head_w1 = (const float*)d_in[34];
  const float* head_b1 = (const float*)d_in[35];
  const float* head_w2 = (const float*)d_in[36];
  const float* head_b2 = (const float*)d_in[37];

  const long BM = (long)B_ROWS * M_MEM;  // 262144

  // ---- workspace layout ----
  char* p = (char*)d_ws;
  __bf16* trunk_w1_b = (__bf16*)bump(p, sizeof(__bf16) * 640 * HID);
  __bf16* trunk_w2_b = (__bf16*)bump(p, sizeof(__bf16) * HID * HID);
  __bf16* mem_w_b    = (__bf16*)bump(p, sizeof(__bf16) * LATENT * HID);
  __bf16* wq_b  = (__bf16*)bump(p, sizeof(__bf16) * HID * HID);
  __bf16* wk_b  = (__bf16*)bump(p, sizeof(__bf16) * HID * HID);
  __bf16* wv_b  = (__bf16*)bump(p, sizeof(__bf16) * HID * HID);
  __bf16* wo_b  = (__bf16*)bump(p, sizeof(__bf16) * HID * HID);
  __bf16* film_w_b  = (__bf16*)bump(p, sizeof(__bf16) * CTX * 2 * HID);
  __bf16* cond_w1_b = (__bf16*)bump(p, sizeof(__bf16) * HID * HID);
  __bf16* cond_w2_b = (__bf16*)bump(p, sizeof(__bf16) * HID * HID);
  __bf16* head_w1_b = (__bf16*)bump(p, sizeof(__bf16) * HID * HID);
  __bf16* head_w2_b = (__bf16*)bump(p, sizeof(__bf16) * HID * LATENT);
  __bf16* csum_b    = (__bf16*)bump(p, sizeof(__bf16) * (size_t)B_ROWS * CTX);
  __bf16* Wkp = (__bf16*)bump(p, sizeof(__bf16) * LATENT * HID);   // mem_w@wk
  __bf16* Wvp = (__bf16*)bump(p, sizeof(__bf16) * LATENT * HID);   // mem_w@wv
  float*  bkp = (float*)bump(p, sizeof(float) * HID);
  float*  bvp = (float*)bump(p, sizeof(float) * HID);
  __bf16* X0  = (__bf16*)bump(p, sizeof(__bf16) * (size_t)B_ROWS * 640);
  __bf16* H1  = (__bf16*)bump(p, sizeof(__bf16) * (size_t)B_ROWS * HID);
  float*  Gf  = (float*)bump(p, sizeof(float)  * (size_t)B_ROWS * HID);
  __bf16* Gb  = (__bf16*)bump(p, sizeof(__bf16) * (size_t)B_ROWS * HID);
  __bf16* SMn = (__bf16*)bump(p, sizeof(__bf16) * (size_t)BM * LATENT);
  __bf16* Kb  = (__bf16*)bump(p, sizeof(__bf16) * (size_t)BM * HID);
  __bf16* Vb  = (__bf16*)bump(p, sizeof(__bf16) * (size_t)BM * HID);
  float*  Qf  = (float*)bump(p, sizeof(float)  * (size_t)B_ROWS * HID);
  __bf16* Sb  = (__bf16*)bump(p, sizeof(__bf16) * (size_t)B_ROWS * HID);
  float*  ATTf = (float*)bump(p, sizeof(float) * (size_t)B_ROWS * HID);
  float*  FILMf = (float*)bump(p, sizeof(float) * (size_t)B_ROWS * 2 * HID);
  float*  COMBf = (float*)bump(p, sizeof(float) * (size_t)B_ROWS * HID);
  __bf16* CXb = (__bf16*)bump(p, sizeof(__bf16) * (size_t)B_ROWS * HID);
  __bf16* H2b = (__bf16*)bump(p, sizeof(__bf16) * (size_t)B_ROWS * HID);
  float*  CONDf = (float*)bump(p, sizeof(float) * (size_t)B_ROWS * HID);
  __bf16* HXb = (__bf16*)bump(p, sizeof(__bf16) * (size_t)B_ROWS * HID);
  __bf16* H3b = (__bf16*)bump(p, sizeof(__bf16) * (size_t)B_ROWS * HID);

  // ---- weight casts to bf16 ----
  cast_bf16(stream, trunk_w1, trunk_w1_b, (size_t)640 * HID);
  cast_bf16(stream, trunk_w2, trunk_w2_b, (size_t)HID * HID);
  cast_bf16(stream, mem_w,    mem_w_b,    (size_t)LATENT * HID);
  cast_bf16(stream, wq, wq_b, (size_t)HID * HID);
  cast_bf16(stream, wk, wk_b, (size_t)HID * HID);
  cast_bf16(stream, wv, wv_b, (size_t)HID * HID);
  cast_bf16(stream, wo, wo_b, (size_t)HID * HID);
  cast_bf16(stream, film_w,  film_w_b,  (size_t)CTX * 2 * HID);
  cast_bf16(stream, cond_w1, cond_w1_b, (size_t)HID * HID);
  cast_bf16(stream, cond_w2, cond_w2_b, (size_t)HID * HID);
  cast_bf16(stream, head_w1, head_w1_b, (size_t)HID * HID);
  cast_bf16(stream, head_w2, head_w2_b, (size_t)HID * LATENT);
  cast_bf16(stream, csum, csum_b, (size_t)B_ROWS * CTX);

  // ---- fold mem projection into K/V weights: Wk' = mem_w@wk, bk' = mem_b@wk + bk ----
  gemm(stream, mem_w_b, wk_b, nullptr, nullptr, nullptr, Wkp, LATENT, HID, HID, 0);
  gemm(stream, mem_w_b, wv_b, nullptr, nullptr, nullptr, Wvp, LATENT, HID, HID, 0);
  fold_bias_kernel<<<(HID + 255) / 256, 256, 0, stream>>>(mem_b, wk, bk, wv, bv, bkp, bvp);

  // ---- trunk ----
  trunk_input_kernel<<<B_ROWS, 256, 0, stream>>>(states, tvals, trunk_ln_g, trunk_ln_b, X0);
  gemm(stream, X0, trunk_w1_b, trunk_b1, nullptr, nullptr, H1, B_ROWS, HID, 640, 1);  // gelu
  gemm(stream, H1, trunk_w2_b, trunk_b2, nullptr, Gf, Gb, B_ROWS, HID, HID, 0);

  // ---- memory path: LN(support) then K,V directly via folded weights ----
  ln_rows_wave_kernel<<<(int)(BM / 8), 256, 0, stream>>>(smem, mem_ln_g, mem_ln_b, SMn, LATENT, BM);
  gemm(stream, SMn, Wkp, bkp, nullptr, nullptr, Kb, (int)BM, HID, LATENT, 0);
  gemm(stream, SMn, Wvp, bvp, nullptr, nullptr, Vb, (int)BM, HID, LATENT, 0);

  // ---- attention ----
  gemm(stream, Gb, wq_b, bq, nullptr, Qf, nullptr, B_ROWS, HID, HID, 0);
  attention_kernel<<<B_ROWS * HEADS, 128, 0, stream>>>(Qf, Kb, Vb, Sb);
  gemm(stream, Sb, wo_b, bo, nullptr, ATTf, nullptr, B_ROWS, HID, HID, 0);

  // ---- FiLM conditioning ----
  gemm(stream, csum_b, film_w_b, film_b, nullptr, FILMf, nullptr, B_ROWS, 2 * HID, CTX, 0);
  film_fuse_kernel<<<B_ROWS, 256, 0, stream>>>(Gf, ATTf, FILMf, pre_ln_g, pre_ln_b,
                                               cond_ln_g, cond_ln_b, COMBf, CXb);
  gemm(stream, CXb, cond_w1_b, cond_b1, nullptr, nullptr, H2b, B_ROWS, HID, HID, 1); // gelu
  gemm(stream, H2b, cond_w2_b, cond_b2, COMBf, CONDf, nullptr, B_ROWS, HID, HID, 0); // +residual

  // ---- head ----
  ln_rows_wave_kernel<<<B_ROWS / 8, 256, 0, stream>>>(CONDf, head_ln_g, head_ln_b, HXb, HID, B_ROWS);
  gemm(stream, HXb, head_w1_b, head_b1, nullptr, nullptr, H3b, B_ROWS, HID, HID, 1); // gelu
  gemm(stream, H3b, head_w2_b, head_b2, nullptr, (float*)d_out, nullptr, B_ROWS, LATENT, HID, 0);
}
